// WindowAttention_58050777973479
// MI455X (gfx1250) — matrix operand order repaired and checked by reading, not compiled, not run on hardware
//
#include <hip/hip_runtime.h>
#include <hip/hip_bf16.h>

// Swin-style shifted-window attention for MI455X (gfx1250, wave32, WMMA).
// Pipeline (all f16 intermediates to halve HBM traffic; memory-bound workload):
//   K1: qkv = roll(x) @ w_qkv   -> scatter q/k/v f16 into window-major ws
//   K2: per-window attention (QK^T, bias, masks, softmax, PV) via WMMA,
//       Q/K tiles staged with GLOBAL_LOAD_ASYNC_TO_LDS (ASYNCcnt path)
//   K3: out = attn @ w_out + b  -> inverse roll fused into store, f32 out

#define WS_SZ     8
#define DISP      4
#define HEADS     12
#define HEAD_DIM  32
#define DIM       384          // HEADS*HEAD_DIM
#define NQKV      1152         // 3*DIM
#define BATCH     16
#define IMG       64
#define SCALE_F   0.17677669529663689f   // 1/sqrt(32)

typedef __attribute__((ext_vector_type(16))) _Float16 v16h;
typedef __attribute__((ext_vector_type(8)))  float    v8f;
typedef int v4i __attribute__((vector_size(16)));   // pointee type of async builtin

#define AS1 __attribute__((address_space(1)))
#define AS3 __attribute__((address_space(3)))

#if __has_builtin(__builtin_amdgcn_global_load_async_to_lds_b128)
#define HAVE_ASYNC_LDS 1
#else
#define HAVE_ASYNC_LDS 0
#endif

__device__ __forceinline__ void async_copy_b128(const void* gsrc, void* ldst) {
#if HAVE_ASYNC_LDS
    __builtin_amdgcn_global_load_async_to_lds_b128(
        (AS1 v4i*)gsrc, (AS3 v4i*)ldst, 0, 0);
#else
    *(uint4*)ldst = *(const uint4*)gsrc;
#endif
}

__device__ __forceinline__ void wait_async_lds() {
#if HAVE_ASYNC_LDS
#if __has_builtin(__builtin_amdgcn_s_wait_asynccnt)
    __builtin_amdgcn_s_wait_asynccnt(0);
#else
    asm volatile("s_wait_asynccnt 0x0" ::: "memory");
#endif
#endif
}

// ---- WMMA fragment helpers (contiguous k order in A and B, wave32) ----------

// A fragment, f16 16x32 (MxK), row-major source with stride ld (halves).
// lane<16: M=lane, K=0..15; lane>=16: M=lane-16, K=16..31 (contiguous per lane)
__device__ __forceinline__ v16h load_a_frag(const _Float16* base, int ld) {
    const int l  = threadIdx.x & 31;
    const int m  = l & 15;
    const int k0 = (l < 16) ? 0 : 16;
    const _Float16* p0 = base + m * ld + k0;
    const _Float16* p1 = p0 + 8;
    v16h a;
#pragma unroll
    for (int i = 0; i < 8; ++i) { a[i] = p0[i]; a[i + 8] = p1[i]; }
    return a;
}

// B fragment, f16 32x16 (KxN), source stored N-major: baseT[n*ld + k].
// lane<16: N=lane, K=0..15; lane>=16: N=lane-16, K=16..31 (contiguous per lane)
__device__ __forceinline__ v16h load_b_frag(const _Float16* baseT, int ld) {
    const int l  = threadIdx.x & 31;
    const int n  = l & 15;
    const int k0 = (l < 16) ? 0 : 16;
    const _Float16* p = baseT + n * ld + k0;
    v16h b;
#pragma unroll
    for (int i = 0; i < 16; ++i) b[i] = p[i];
    return b;
}

__device__ __forceinline__ v8f wmma16(v16h a, v16h b, v8f c) {
    return __builtin_amdgcn_wmma_f32_16x16x32_f16(
        /*neg_a=*/false, a, /*neg_b=*/false, b,
        /*c_mod=*/(short)0, c, /*reuse_a=*/false, /*reuse_b=*/false);
}

// ---- Kernel 1: fused cyclic-shift + QKV GEMM + window-scatter (f16 out) ----
// grid (1024, 9), block 256 (8 waves, 2x4), tile 64(M) x 128(N), K-step 32
__global__ __launch_bounds__(256)
void qkv_kernel(const float* __restrict__ x, const float* __restrict__ w_qkv,
                _Float16* __restrict__ q_ws, _Float16* __restrict__ k_ws,
                _Float16* __restrict__ v_ws) {
    __shared__ _Float16 As[64][32];    // [m][k]
    __shared__ _Float16 Bt[128][32];   // [n][k] (transposed)

    const int mblk = blockIdx.x;
    const int nblk = blockIdx.y;
    const int tid  = threadIdx.x;
    const int lane = tid & 31;
    const int wave = tid >> 5;
    const int wm   = wave >> 2;        // 0..1  -> 32-row strip
    const int wn   = wave & 3;         // 0..3  -> 32-col strip

    // A-load mapping: 64 rows x 32 k, 8 floats/thread; shift fused in address
    const int ar = tid >> 2;
    const int ak = (tid & 3) * 8;
    const int mg = mblk * 64 + ar;
    const int bb = mg >> 12;
    const int hh = (mg >> 6) & 63;
    const int ww = mg & 63;
    const float* xrow = x + (((size_t)bb * IMG + ((hh + DISP) & 63)) * IMG
                             + ((ww + DISP) & 63)) * DIM;
    // B-load mapping: 32 k-rows x 128 cols, 16 cols/thread
    const int bkr = tid >> 3;
    const int bnc = (tid & 7) * 16;
    const float* bbase = w_qkv + (size_t)nblk * 128 + bnc;

    v8f acc[2][2] = {{{0.f}, {0.f}}, {{0.f}, {0.f}}};

    for (int k0 = 0; k0 < DIM; k0 += 32) {
        __syncthreads();
        // stage A (f32 -> f16)
        {
            const float4* src = (const float4*)(xrow + k0 + ak);
            float4 f0 = src[0], f1 = src[1];
            _Float16* d = &As[ar][ak];
            d[0] = (_Float16)f0.x; d[1] = (_Float16)f0.y;
            d[2] = (_Float16)f0.z; d[3] = (_Float16)f0.w;
            d[4] = (_Float16)f1.x; d[5] = (_Float16)f1.y;
            d[6] = (_Float16)f1.z; d[7] = (_Float16)f1.w;
            if (k0 + 32 < DIM) __builtin_prefetch(xrow + k0 + 32 + ak, 0, 1);
        }
        // stage B transposed (f32 -> f16)
        {
            const float* brow = bbase + (size_t)(k0 + bkr) * NQKV;
#pragma unroll
            for (int j = 0; j < 16; ++j) Bt[bnc + j][bkr] = (_Float16)brow[j];
            if (k0 + 32 < DIM) __builtin_prefetch(bbase + (size_t)(k0 + 32 + bkr) * NQKV, 0, 1);
        }
        __syncthreads();

        v16h af[2], bf[2];
#pragma unroll
        for (int rt = 0; rt < 2; ++rt) af[rt] = load_a_frag(&As[wm * 32 + rt * 16][0], 32);
#pragma unroll
        for (int ct = 0; ct < 2; ++ct) bf[ct] = load_b_frag(&Bt[wn * 32 + ct * 16][0], 32);
#pragma unroll
        for (int rt = 0; rt < 2; ++rt)
#pragma unroll
            for (int ct = 0; ct < 2; ++ct)
                acc[rt][ct] = wmma16(af[rt], bf[ct], acc[rt][ct]);
    }

    // epilogue: scatter f16 into window-major [b][head][win][tok][d]
    const int nl = lane & 15;
    const int mb = (lane < 16) ? 0 : 8;
#pragma unroll
    for (int rt = 0; rt < 2; ++rt)
#pragma unroll
        for (int ct = 0; ct < 2; ++ct) {
            const int ng    = nblk * 128 + wn * 32 + ct * 16 + nl;
            const int which = ng / DIM;
            const int col   = ng % DIM;
            const int head  = col >> 5;
            const int dd    = col & 31;
            _Float16* dst = (which == 0) ? q_ws : (which == 1) ? k_ws : v_ws;
#pragma unroll
            for (int r = 0; r < 8; ++r) {
                const int m  = mblk * 64 + wm * 32 + rt * 16 + mb + r;
                const int b2 = m >> 12;
                const int h2 = (m >> 6) & 63;
                const int w2 = m & 63;
                const int win = ((h2 >> 3) << 3) | (w2 >> 3);
                const int tok = ((h2 & 7) << 3) | (w2 & 7);
                const size_t idx =
                    ((((size_t)b2 * HEADS + head) * 64 + win) * 64 + tok) * HEAD_DIM + dd;
                dst[idx] = (_Float16)acc[rt][ct][r];
            }
        }
}

// ---- Kernel 2: window attention, one wave per window, 2 windows/block ------
__global__ __launch_bounds__(64)
void attn_kernel(const _Float16* __restrict__ q_ws, const _Float16* __restrict__ k_ws,
                 const _Float16* __restrict__ v_ws, const float* __restrict__ pe,
                 _Float16* __restrict__ a_ws) {
    __shared__ _Float16 Qs[2][64][32];   // [tok][d]
    __shared__ _Float16 Ks[2][64][32];   // [tok][d] == B layout for Q*K^T
    __shared__ _Float16 Vt[2][32][64];   // [d][tok] == B layout for P*V
    __shared__ float    Sb[2][64][64];   // scores f32; P written f16 in place
    __shared__ float    pe_s[225];

    const int tid  = threadIdx.x;
    const int lane = tid & 31;
    const int wv   = tid >> 5;
    for (int i = tid; i < 225; i += 64) pe_s[i] = pe[i];

    const int widx = blockIdx.x * 2 + wv;           // = ((b*12+head)*64+win)
    const int bb   = widx / (HEADS * 64);
    const int rem  = widx % (HEADS * 64);
    const int head = rem >> 6;
    const int win  = rem & 63;
    const int wr   = win >> 3, wc = win & 7;
    const size_t base = (size_t)widx * (64 * HEAD_DIM);

    // stage Q,K via async memory->LDS DMA (pure f16 copies, no conversion);
    // V transposed through VGPRs (needs data rearrangement anyway)
    {
        const uint4* qg = (const uint4*)(q_ws + base);
        const uint4* kg = (const uint4*)(k_ws + base);
        uint4* qs = (uint4*)&Qs[wv][0][0];
        uint4* ks = (uint4*)&Ks[wv][0][0];
#pragma unroll
        for (int i0 = 0; i0 < 256; i0 += 32) {
            const int i = i0 + lane;
            async_copy_b128(qg + i, qs + i);
            async_copy_b128(kg + i, ks + i);
        }
        for (int t = lane; t < 64; t += 32) {
            const _Float16* vrow = v_ws + base + (size_t)t * HEAD_DIM;
#pragma unroll
            for (int d2 = 0; d2 < 32; ++d2) Vt[wv][d2][t] = vrow[d2];
        }
    }
    wait_async_lds();     // drain ASYNCcnt; barriers do not cover async DMA
    __syncthreads();

    // S = Q K^T * scale + bias + masks  (K=32 -> single WMMA per 16x16 tile)
    const int nl = lane & 15;
    const int mb = (lane < 16) ? 0 : 8;
#pragma unroll
    for (int rt = 0; rt < 4; ++rt) {
        v16h a = load_a_frag(&Qs[wv][rt * 16][0], 32);
#pragma unroll
        for (int ct = 0; ct < 4; ++ct) {
            v16h b = load_b_frag(&Ks[wv][ct * 16][0], 32);
            v8f c = {0.f};
            c = wmma16(a, b, c);
            const int j  = ct * 16 + nl;
            const int yj = j >> 3, xj = j & 7;
#pragma unroll
            for (int r = 0; r < 8; ++r) {
                const int i  = rt * 16 + mb + r;
                const int yi = i >> 3, xi = i & 7;
                float v = c[r] * SCALE_F + pe_s[(yi - yj + 7) * 15 + (xi - xj + 7)];
                if (wr == 7 && ((i >= 32) != (j >= 32))) v = -1e30f;
                if (wc == 7 && ((xi >= 4) != (xj >= 4))) v = -1e30f;
                Sb[wv][i][j] = v;
            }
        }
    }
    __syncthreads();

    // softmax rows (lane, lane+32); write P f16 in place (row stride 128 halves)
#pragma unroll
    for (int rr = 0; rr < 2; ++rr) {
        const int i = lane + rr * 32;
        float vals[64];
        float mx = -1e30f;
#pragma unroll
        for (int j = 0; j < 64; ++j) { vals[j] = Sb[wv][i][j]; mx = fmaxf(mx, vals[j]); }
        float s = 0.f;
#pragma unroll
        for (int j = 0; j < 64; ++j) { vals[j] = __expf(vals[j] - mx); s += vals[j]; }
        const float inv = 1.f / s;
        _Float16* prow = (_Float16*)&Sb[wv][i][0];
#pragma unroll
        for (int j = 0; j < 64; ++j) prow[j] = (_Float16)(vals[j] * inv);
    }
    __syncthreads();

    // O = P V  (K=64 -> 2 WMMA steps), store f16 row-major [m][384]
#pragma unroll
    for (int rt = 0; rt < 4; ++rt) {
#pragma unroll
        for (int ct = 0; ct < 2; ++ct) {
            v8f c = {0.f};
#pragma unroll
            for (int kb = 0; kb < 2; ++kb) {
                v16h a = load_a_frag((const _Float16*)&Sb[wv][rt * 16][0] + kb * 32, 128);
                v16h b = load_b_frag(&Vt[wv][ct * 16][0] + kb * 32, 64);
                c = wmma16(a, b, c);
            }
            const int d = ct * 16 + nl;
#pragma unroll
            for (int r = 0; r < 8; ++r) {
                const int t  = rt * 16 + mb + r;
                const int h  = wr * 8 + (t >> 3);
                const int w  = wc * 8 + (t & 7);
                const size_t m = ((size_t)bb * IMG + h) * IMG + w;
                a_ws[m * DIM + head * HEAD_DIM + d] = (_Float16)c[r];
            }
        }
    }
}

// ---- Kernel 3: output projection + bias + inverse cyclic shift (f32 out) ---
// grid (1024, 3), block 256, tile 64 x 128, K-step 32
__global__ __launch_bounds__(256)
void out_kernel(const _Float16* __restrict__ a_ws, const float* __restrict__ w_out,
                const float* __restrict__ b_out, float* __restrict__ out) {
    __shared__ _Float16 As[64][32];
    __shared__ _Float16 Bt[128][32];

    const int mblk = blockIdx.x;
    const int nblk = blockIdx.y;
    const int tid  = threadIdx.x;
    const int lane = tid & 31;
    const int wave = tid >> 5;
    const int wm   = wave >> 2;
    const int wn   = wave & 3;

    const int ar = tid >> 2;
    const int ak = (tid & 3) * 8;
    const _Float16* arow = a_ws + (size_t)(mblk * 64 + ar) * DIM;
    const int bkr = tid >> 3;
    const int bnc = (tid & 7) * 16;
    const float* bbase = w_out + (size_t)nblk * 128 + bnc;

    v8f acc[2][2] = {{{0.f}, {0.f}}, {{0.f}, {0.f}}};

    for (int k0 = 0; k0 < DIM; k0 += 32) {
        __syncthreads();
        // A tile is already f16: pure async memory->LDS copy
        async_copy_b128(arow + k0 + ak, &As[ar][ak]);
        {
            const float* brow = bbase + (size_t)(k0 + bkr) * DIM;
#pragma unroll
            for (int j = 0; j < 16; ++j) Bt[bnc + j][bkr] = (_Float16)brow[j];
            if (k0 + 32 < DIM) {
                __builtin_prefetch(arow + k0 + 32 + ak, 0, 1);
                __builtin_prefetch(bbase + (size_t)(k0 + 32 + bkr) * DIM, 0, 1);
            }
        }
        wait_async_lds();
        __syncthreads();

        v16h af[2], bf[2];
#pragma unroll
        for (int rt = 0; rt < 2; ++rt) af[rt] = load_a_frag(&As[wm * 32 + rt * 16][0], 32);
#pragma unroll
        for (int ct = 0; ct < 2; ++ct) bf[ct] = load_b_frag(&Bt[wn * 32 + ct * 16][0], 32);
#pragma unroll
        for (int rt = 0; rt < 2; ++rt)
#pragma unroll
            for (int ct = 0; ct < 2; ++ct)
                acc[rt][ct] = wmma16(af[rt], bf[ct], acc[rt][ct]);
    }

    const int nl = lane & 15;
    const int mb = (lane < 16) ? 0 : 8;
#pragma unroll
    for (int rt = 0; rt < 2; ++rt)
#pragma unroll
        for (int ct = 0; ct < 2; ++ct) {
            const int ng = nblk * 128 + wn * 32 + ct * 16 + nl;
            const float bias = b_out[ng];
#pragma unroll
            for (int r = 0; r < 8; ++r) {
                const int m  = mblk * 64 + wm * 32 + rt * 16 + mb + r;
                const int b2 = m >> 12;
                const int h2 = ((m >> 6) & 63);
                const int w2 = m & 63;
                const int ho = (h2 + DISP) & 63;     // inverse roll fused here
                const int wo = (w2 + DISP) & 63;
                out[(((size_t)b2 * IMG + ho) * IMG + wo) * DIM + ng] = acc[rt][ct][r] + bias;
            }
        }
}

// ---- launch ----------------------------------------------------------------
extern "C" void kernel_launch(void* const* d_in, const int* in_sizes, int n_in,
                              void* d_out, int out_size, void* d_ws, size_t ws_size,
                              hipStream_t stream) {
    (void)in_sizes; (void)n_in; (void)out_size; (void)ws_size;
    const float* x     = (const float*)d_in[0];
    const float* w_qkv = (const float*)d_in[1];
    const float* pe    = (const float*)d_in[2];
    const float* w_out = (const float*)d_in[3];
    const float* b_out = (const float*)d_in[4];
    float* out = (float*)d_out;

    const size_t QKV_ELEMS = (size_t)BATCH * HEADS * 64 * 64 * HEAD_DIM;  // 25,165,824
    _Float16* q_ws = (_Float16*)d_ws;
    _Float16* k_ws = q_ws + QKV_ELEMS;
    _Float16* v_ws = k_ws + QKV_ELEMS;
    _Float16* a_ws = v_ws + QKV_ELEMS;   // 65536 x 384 f16

    qkv_kernel<<<dim3(1024, 9), 256, 0, stream>>>(x, w_qkv, q_ws, k_ws, v_ws);
    attn_kernel<<<dim3(6144), 64, 0, stream>>>(q_ws, k_ws, v_ws, pe, a_ws);
    out_kernel<<<dim3(1024, 3), 256, 0, stream>>>(a_ws, w_out, b_out, out);
}
